// MiniModel_7980049236542
// MI455X (gfx1250) — matrix operand order repaired and checked by reading, not compiled, not run on hardware
//
#include <hip/hip_runtime.h>
#include <hip/hip_bf16.h>
#include <math.h>

// Problem constants (from reference)
#define BB    256
#define TT    128
#define DD    512
#define DEPTHN 4
#define SS    256
#define BNK   51     // BN = max(512//10, 8)
#define MB    16     // batches per workgroup (== WMMA M)
#define NWAVE 16     // waves per workgroup (wave32)
#define NTHR  (NWAVE * 32)

typedef __bf16 v16bf __attribute__((ext_vector_type(16)));
typedef float  v8f   __attribute__((ext_vector_type(8)));
typedef unsigned int uint4v __attribute__((ext_vector_type(4)));
typedef float  float4v __attribute__((ext_vector_type(4)));

union V16U { v16bf v; uint4v u[2]; };

__device__ __forceinline__ unsigned short f2bf(float f) {
  unsigned int u = __float_as_uint(f);
  unsigned int r = u + 0x7FFFu + ((u >> 16) & 1u);   // round-to-nearest-even
  return (unsigned short)(r >> 16);
}

__device__ __forceinline__ float gelu_exact(float x) {
  return 0.5f * x * (1.0f + erff(x * 0.70710678118654752440f));
}

// ---------------------------------------------------------------- prep kernels

__global__ void prep_weights(const float* __restrict__ W1, const float* __restrict__ W2,
                             unsigned short* __restrict__ Wt1, unsigned short* __restrict__ Wt2) {
  // Wt[i][n][k] = bf16(W[i][k][n])  (column-major so B-fragments are contiguous in K)
  const size_t n = (size_t)DEPTHN * DD * DD;
  for (size_t e = (size_t)blockIdx.x * blockDim.x + threadIdx.x; e < n;
       e += (size_t)gridDim.x * blockDim.x) {
    size_t i   = e / ((size_t)DD * DD);
    size_t rem = e % ((size_t)DD * DD);
    size_t nn  = rem / DD;
    size_t kk  = rem % DD;
    size_t src = i * DD * DD + kk * DD + nn;
    Wt1[e] = f2bf(W1[src]);
    Wt2[e] = f2bf(W2[src]);
  }
}

__global__ void prep_mem(const float4v* __restrict__ src, float4v* __restrict__ dst) {
  const size_t n4 = (size_t)BB * SS * DD / 4;
  for (size_t e = (size_t)blockIdx.x * blockDim.x + threadIdx.x; e < n4;
       e += (size_t)gridDim.x * blockDim.x)
    dst[e] = src[e];
}

__global__ void zero_margins(float* __restrict__ m) {
  if (threadIdx.x < TT) m[threadIdx.x] = 0.0f;
}

// ---------------------------------------------------------------- main kernel

__global__ __launch_bounds__(NTHR)
void minimodel_main(const float* __restrict__ x,   const float* __restrict__ Wi,
                    const float* __restrict__ bi,
                    const float* __restrict__ b1g, const float* __restrict__ b2g,
                    const float* __restrict__ Wb1, const float* __restrict__ bb1,
                    const float* __restrict__ Wb2, const float* __restrict__ bb2,
                    const float* __restrict__ Wb3, const float* __restrict__ bb3,
                    const float* __restrict__ zg,
                    const float* __restrict__ Wo,  const float* __restrict__ bo,
                    const unsigned short* __restrict__ Wt1,
                    const unsigned short* __restrict__ Wt2,
                    float* __restrict__ memws,
                    float* __restrict__ outs,      // [B][T]
                    float* __restrict__ margins) { // [T] (pre-zeroed, atomic accum)
  __shared__ float          Hid[MB][DD];                 // 32 KB persistent hidden state
  __shared__ __align__(16) unsigned short Abuf[MB][DD];  // 16 KB bf16 A staging
  __shared__ int            slotL[MB];
  __shared__ float          marginL[MB];
  __shared__ float          T1[MB][BNK + 1];
  __shared__ float          T2[MB][BNK + 1];

  const int tid  = threadIdx.x;
  const int lane = tid & 31;
  const int wave = tid >> 5;
  const int gb0  = blockIdx.x * MB;

  const int rowA  = lane & 15;
  const int koffA = (lane < 16) ? 0 : 8;    // A-frag K layout (ISA 16-bit A 16x32)
  const int koffB = (lane < 16) ? 0 : 8;    // B-frag K layout
  const int nbase = wave * 32;              // each wave owns 32 output columns
  const int colT0 = nbase + (lane & 15);
  const int colT1 = colT0 + 16;
  const int rbase = (lane < 16) ? 0 : 8;    // C/D row layout

  const float gate = 1.0f / (1.0f + expf(-zg[0]));

  for (int e = tid; e < MB * DD; e += NTHR) Hid[e >> 9][e & 511] = 0.0f;
  __syncthreads();

  for (int t = 0; t < TT; ++t) {
    // ---- phase 0: hidden += x_t * Wi + bi
    for (int e = tid; e < MB * DD; e += NTHR) {
      int b = e >> 9, d = e & 511;
      Hid[b][d] += x[(size_t)(gb0 + b) * TT + t] * Wi[d] + bi[d];
    }
    __syncthreads();

    // ---- DEPTH residual MLP layers via WMMA (M=16, N=512, K=512)
    for (int i = 0; i < DEPTHN; ++i) {
      // stage hidden -> bf16 A buffer
      for (int e = tid; e < MB * DD; e += NTHR)
        Abuf[e >> 9][e & 511] = f2bf(Hid[e >> 9][e & 511]);
      __syncthreads();

      // GEMM1: h1 = Abuf @ W1[i]
      v8f acc0 = {}; v8f acc1 = {};
      {
        const unsigned short* B0 = Wt1 + ((size_t)i * DD + colT0) * DD + koffB;
        const unsigned short* B1 = Wt1 + ((size_t)i * DD + colT1) * DD + koffB;
        for (int k0 = 0; k0 < DD; k0 += 32) {
          V16U a, bA, bB;
          a.u[0]  = *(const uint4v*)&Abuf[rowA][k0 + koffA];
          a.u[1]  = *(const uint4v*)&Abuf[rowA][k0 + koffA + 16];
          bA.u[0] = *(const uint4v*)(B0 + k0);
          bA.u[1] = *(const uint4v*)(B0 + k0 + 16);
          bB.u[0] = *(const uint4v*)(B1 + k0);
          bB.u[1] = *(const uint4v*)(B1 + k0 + 16);
          if (k0 + 64 < DD) __builtin_prefetch(B0 + k0 + 64, 0, 1);
          acc0 = __builtin_amdgcn_wmma_f32_16x16x32_bf16(false, a.v, false, bA.v,
                                                         (short)0, acc0, false, false);
          acc1 = __builtin_amdgcn_wmma_f32_16x16x32_bf16(false, a.v, false, bB.v,
                                                         (short)0, acc1, false, false);
        }
      }
      __syncthreads();  // all waves finished reading Abuf; accs live in VGPRs
      {
        const float c0b = b1g[i * DD + colT0];
        const float c1b = b1g[i * DD + colT1];
#pragma unroll
        for (int r = 0; r < 8; ++r) {
          Abuf[rbase + r][colT0] = f2bf(gelu_exact(acc0[r] + c0b));
          Abuf[rbase + r][colT1] = f2bf(gelu_exact(acc1[r] + c1b));
        }
      }
      __syncthreads();

      // GEMM2: hidden += gelu(h1) @ W2[i] + b2[i]
      acc0 = (v8f){}; acc1 = (v8f){};
      {
        const unsigned short* B0 = Wt2 + ((size_t)i * DD + colT0) * DD + koffB;
        const unsigned short* B1 = Wt2 + ((size_t)i * DD + colT1) * DD + koffB;
        for (int k0 = 0; k0 < DD; k0 += 32) {
          V16U a, bA, bB;
          a.u[0]  = *(const uint4v*)&Abuf[rowA][k0 + koffA];
          a.u[1]  = *(const uint4v*)&Abuf[rowA][k0 + koffA + 16];
          bA.u[0] = *(const uint4v*)(B0 + k0);
          bA.u[1] = *(const uint4v*)(B0 + k0 + 16);
          bB.u[0] = *(const uint4v*)(B1 + k0);
          bB.u[1] = *(const uint4v*)(B1 + k0 + 16);
          acc0 = __builtin_amdgcn_wmma_f32_16x16x32_bf16(false, a.v, false, bA.v,
                                                         (short)0, acc0, false, false);
          acc1 = __builtin_amdgcn_wmma_f32_16x16x32_bf16(false, a.v, false, bB.v,
                                                         (short)0, acc1, false, false);
        }
      }
      {
        const float c0b = b2g[i * DD + colT0];
        const float c1b = b2g[i * DD + colT1];
#pragma unroll
        for (int r = 0; r < 8; ++r) {
          Hid[rbase + r][colT0] += acc0[r] + c0b;
          Hid[rbase + r][colT1] += acc1[r] + c1b;
        }
      }
      __syncthreads();
    }

    // ---- scores + top-2 (wave w handles batch w; mem is L2-resident)
    {
      const int b = wave;
      const float* hrow  = &Hid[b][0];
      const float* mbase = memws + (size_t)(gb0 + b) * SS * DD;
      float acc[8];
#pragma unroll
      for (int j = 0; j < 8; ++j) acc[j] = 0.0f;
      for (int d0 = 0; d0 < DD; d0 += 4) {
        float4v hv = *(const float4v*)&hrow[d0];
#pragma unroll
        for (int j = 0; j < 8; ++j) {
          const int s = lane + 32 * j;
          float4v mv = *(const float4v*)&mbase[(size_t)s * DD + d0];
          acc[j] += hv[0] * mv[0] + hv[1] * mv[1] + hv[2] * mv[2] + hv[3] * mv[3];
        }
      }
      float v1 = -__builtin_inff(), v2 = -__builtin_inff();
      int   i1 = 0x7FFFFFFF,        i2 = 0x7FFFFFFF;
#pragma unroll
      for (int j = 0; j < 8; ++j) {
        const float v = acc[j]; const int s = lane + 32 * j;
        if (v > v1 || (v == v1 && s < i1)) { v2 = v1; i2 = i1; v1 = v; i1 = s; }
        else if (v > v2 || (v == v2 && s < i2)) { v2 = v; i2 = s; }
      }
      for (int off = 16; off > 0; off >>= 1) {
        float ov1 = __shfl_xor(v1, off, 32); int oi1 = __shfl_xor(i1, off, 32);
        float ov2 = __shfl_xor(v2, off, 32); int oi2 = __shfl_xor(i2, off, 32);
        if (ov1 > v1 || (ov1 == v1 && oi1 < i1)) {
          float nv2; int ni2;
          if (v1 > ov2 || (v1 == ov2 && i1 < oi2)) { nv2 = v1; ni2 = i1; }
          else                                     { nv2 = ov2; ni2 = oi2; }
          v1 = ov1; i1 = oi1; v2 = nv2; i2 = ni2;
        } else if (ov1 > v2 || (ov1 == v2 && oi1 < i2)) { v2 = ov1; i2 = oi1; }
      }
      if (lane == 0) { slotL[b] = i1; marginL[b] = v1 - v2; }
    }
    __syncthreads();
    if (tid == 0) {
      float s = 0.0f;
      for (int b = 0; b < MB; ++b) s += marginL[b];
      atomicAdd(&margins[t], s * (1.0f / (float)BB));
    }

    // ---- bottleneck MLP on context rows (pre-update mem)
    for (int o = tid; o < MB * BNK; o += NTHR) {
      const int b = o / BNK, n = o % BNK;
      const float* ctx = memws + ((size_t)(gb0 + b) * SS + slotL[b]) * DD;
      float a = bb1[n];
      for (int d = 0; d < DD; ++d) a += ctx[d] * Wb1[d * BNK + n];
      T1[b][n] = gelu_exact(a);
    }
    __syncthreads();
    for (int o = tid; o < MB * BNK; o += NTHR) {
      const int b = o / BNK, n = o % BNK;
      float a = bb2[n];
      for (int m = 0; m < BNK; ++m) a += T1[b][m] * Wb2[m * BNK + n];
      T2[b][n] = gelu_exact(a);
    }
    __syncthreads();
    for (int e = tid; e < MB * DD; e += NTHR) {
      const int b = e >> 9, d = e & 511;
      float z = bb3[d];
      for (int m = 0; m < BNK; ++m) z += T2[b][m] * Wb3[m * DD + d];
      Hid[b][d] += gate * z;
    }
    __syncthreads();

    // ---- mem[b, slot] = hidden ; out[b,t] = hidden . Wo + bo
    for (int e = tid; e < MB * DD; e += NTHR) {
      const int b = e >> 9, d = e & 511;
      memws[((size_t)(gb0 + b) * SS + slotL[b]) * DD + d] = Hid[b][d];
    }
    {
      const int b = wave;
      float p = 0.0f;
#pragma unroll
      for (int j = 0; j < 16; ++j) { const int d = lane + 32 * j; p += Hid[b][d] * Wo[d]; }
      for (int off = 16; off > 0; off >>= 1) p += __shfl_xor(p, off, 32);
      if (lane == 0) outs[(size_t)(gb0 + b) * TT + t] = p + bo[0];
    }
    __threadfence_block();  // make mem writeback visible to next step's score loads
    __syncthreads();
  }
}

// ---------------------------------------------------------------- launch

extern "C" void kernel_launch(void* const* d_in, const int* in_sizes, int n_in,
                              void* d_out, int out_size, void* d_ws, size_t ws_size,
                              hipStream_t stream) {
  (void)in_sizes; (void)n_in; (void)out_size; (void)ws_size;
  const float* x    = (const float*)d_in[0];
  const float* mem0 = (const float*)d_in[1];
  const float* Wi   = (const float*)d_in[2];
  const float* bi   = (const float*)d_in[3];
  const float* W1   = (const float*)d_in[4];
  const float* b1   = (const float*)d_in[5];
  const float* W2   = (const float*)d_in[6];
  const float* b2   = (const float*)d_in[7];
  const float* Wb1  = (const float*)d_in[8];
  const float* bb1  = (const float*)d_in[9];
  const float* Wb2  = (const float*)d_in[10];
  const float* bb2  = (const float*)d_in[11];
  const float* Wb3  = (const float*)d_in[12];
  const float* bb3  = (const float*)d_in[13];
  const float* zg   = (const float*)d_in[14];
  const float* Wo   = (const float*)d_in[15];
  const float* bo   = (const float*)d_in[16];

  float* outs    = (float*)d_out;
  float* margins = outs + (size_t)BB * TT;

  unsigned char* ws = (unsigned char*)d_ws;
  unsigned short* Wt1 = (unsigned short*)ws;                                   // 2 MB
  unsigned short* Wt2 = (unsigned short*)(ws + (size_t)DEPTHN * DD * DD * 2);  // 2 MB
  float*          mw  = (float*)(ws + (size_t)DEPTHN * DD * DD * 4);           // 134 MB

  prep_weights<<<512, 256, 0, stream>>>(W1, W2, Wt1, Wt2);
  prep_mem<<<2048, 256, 0, stream>>>((const float4v*)mem0, (float4v*)mw);
  zero_margins<<<1, 128, 0, stream>>>(margins);
  minimodel_main<<<BB / MB, NTHR, 0, stream>>>(x, Wi, bi, b1, b2, Wb1, bb1, Wb2, bb2,
                                               Wb3, bb3, zg, Wo, bo, Wt1, Wt2, mw,
                                               outs, margins);
}
